// MambaModel_68547678044539
// MI455X (gfx1250) — matrix operand order repaired and checked by reading, not compiled, not run on hardware
//
#include <hip/hip_runtime.h>
#include <hip/hip_bf16.h>

// ---------------------------------------------------------------------------
// Model dims (fixed by the reference)
// ---------------------------------------------------------------------------
#define D_MODEL  1024
#define D_INNER  2048
#define D_STATE  16
#define D_CONV   4
#define DT_RANK  64
#define OUT_DIM  512
#define BATCH    4
#define SEQ      1024
#define NTOK     (BATCH * SEQ)          // 4096 tokens

// ---------------------------------------------------------------------------
// WMMA fragment types (gfx1250, wave32)
// ---------------------------------------------------------------------------
typedef __attribute__((ext_vector_type(16))) __bf16 v16bf;
typedef __attribute__((ext_vector_type(8)))  float  v8f;

union Frag16 {                 // 16 bf16 = 8 VGPRs
    v16bf v;
    unsigned int u[8];
};

static __device__ __forceinline__ v8f wmma_bf16(const Frag16& a, const Frag16& b, v8f c) {
    return __builtin_amdgcn_wmma_f32_16x16x32_bf16(
               false, a.v, false, b.v, (short)0, c, false, false);
}

static __device__ __forceinline__ float softplus_f(float x) {
    return (x > 20.0f) ? x : log1pf(__expf(x));
}
static __device__ __forceinline__ float silu_f(float x) {
    return x * (1.0f / (1.0f + __expf(-x)));
}

// Low 32 bits of a generic pointer to __shared__ data are the hardware LDS
// address (ISA 10.2).
static __device__ __forceinline__ unsigned lds_addr(const void* p) {
    return (unsigned)(size_t)p;
}

// CDNA5 async global->LDS DMA (ASYNCcnt-tracked), 16B per lane.
static __device__ __forceinline__ void async_copy_b128(unsigned lds_off, const void* gaddr) {
    asm volatile("global_load_async_to_lds_b128 %0, %1, off"
                 :: "v"(lds_off), "v"(gaddr) : "memory");
}
static __device__ __forceinline__ void wait_asynccnt0() {
    asm volatile("s_wait_asynccnt 0x0" ::: "memory");
}

// ---------------------------------------------------------------------------
// Weight prep: W[K,N] fp32  ->  Wt[N,K] bf16  (one-time per call)
// ---------------------------------------------------------------------------
__global__ void __launch_bounds__(256)
transpose_convert_kernel(const float* __restrict__ W, __bf16* __restrict__ Wt,
                         int K, int N) {
    size_t gid = (size_t)blockIdx.x * 256 + threadIdx.x;
    if (gid >= (size_t)K * N) return;
    int n = (int)(gid / K);
    int k = (int)(gid - (size_t)n * K);
    Wt[gid] = (__bf16)W[(size_t)k * N + n];
}

// ---------------------------------------------------------------------------
// Embedding gather + sqrt(d_model) scale -> bf16 (GEMM-A operand only)
// ---------------------------------------------------------------------------
__global__ void __launch_bounds__(256)
embed_kernel(const int* __restrict__ src, const float* __restrict__ emb,
             __bf16* __restrict__ x_bf) {
    size_t gid = (size_t)blockIdx.x * 256 + threadIdx.x;   // NTOK*D_MODEL
    int t = (int)(gid >> 10);
    int d = (int)(gid & (D_MODEL - 1));
    x_bf[gid] = (__bf16)(emb[(size_t)src[t] * D_MODEL + d] * 32.0f);
}

// ---------------------------------------------------------------------------
// Depthwise causal conv1d (width 4) + bias + SiLU -> fp32 (scan) + bf16 (GEMM)
// ---------------------------------------------------------------------------
__global__ void __launch_bounds__(256)
conv_silu_kernel(const float* __restrict__ xz, const float* __restrict__ conv_w,
                 const float* __restrict__ conv_b,
                 float* __restrict__ xs, __bf16* __restrict__ xs_bf) {
    size_t gid = (size_t)blockIdx.x * 256 + threadIdx.x;   // NTOK*D_INNER
    int d = (int)(gid & (D_INNER - 1));
    int t = (int)(gid >> 11);
    int l = t & (SEQ - 1);
    float s = conv_b[d];
#pragma unroll
    for (int j = 0; j < D_CONV; ++j) {
        int lj = l - (D_CONV - 1) + j;
        if (lj >= 0)
            s += conv_w[d * D_CONV + j] *
                 xz[(size_t)(t - (D_CONV - 1) + j) * (2 * D_INNER) + d];
    }
    float v = silu_f(s);
    xs[gid] = v;
    xs_bf[gid] = (__bf16)v;
}

// ---------------------------------------------------------------------------
// bf16 WMMA GEMM, double-buffered async global->LDS staging.
//   C[M,N] = A[M,K](lda, bf16) * Bt[N,K](bf16, pre-transposed)
//   Block: 256 thr = 8 waves; block tile 128x128, K-step 32.
//   Wave tile 32x64: 2 A frags x 4 B frags -> 8 v_wmma per K-step.
//   Stage k+1 is DMA'd (ASYNCcnt) into the other LDS buffer while stage k
//   computes; s_wait_asynccnt + split barriers order DMA vs ds_load.
//   LDS rows: 80B stride (16B-aligned for async b128, odd 16B-chunk stride).
//   flags: bit0 = +bias[col], bit1 = softplus. Cf / Cbf optional outputs.
// ---------------------------------------------------------------------------
#define BMT 128
#define BNT 128
#define BKT 32
#define ASTR 40                 // halfwords per LDS row

__global__ void __launch_bounds__(256)
gemm_wmma_async(const __bf16* __restrict__ A, int lda,
                const __bf16* __restrict__ Bt,
                float* __restrict__ Cf, __bf16* __restrict__ Cbf,
                const float* __restrict__ bias,
                int M, int N, int K, int flags) {
    __shared__ unsigned short sA[2][BMT * ASTR];   // 2 x 10240 B
    __shared__ unsigned short sB[2][BNT * ASTR];   // 2 x 10240 B

    const int tid  = threadIdx.x;
    const int lane = tid & 31;
    const int wave = tid >> 5;
    const int rowBase = blockIdx.y * BMT;
    const int colBase = blockIdx.x * BNT;
    const int wr = (wave & 3) * 32;             // wave rows: wr..wr+31
    const int wc = (wave >> 2) * 64;            // wave cols: wc..wc+63
    const int fm   = lane & 15;
    const int half = lane >> 4;

    // 16B-chunk ownership: rows r0 (0..63) and r1 (64..127), 4 chunks/row.
    const int r0  = tid >> 2;
    const int r1  = r0 + 64;
    const int ch8  = (tid & 3) * 8;             // element offset in row
    const int ch16 = (tid & 3) * 16;            // byte offset in row
    const int bn0c = (colBase + r0 < N) ? colBase + r0 : N - 1;  // clamp: EXEC stays uniform
    const int bn1c = (colBase + r1 < N) ? colBase + r1 : N - 1;

    unsigned lA0[2], lA1[2], lB0[2], lB1[2];
#pragma unroll
    for (int p = 0; p < 2; ++p) {
        lA0[p] = lds_addr(&sA[p][r0 * ASTR]) + ch16;
        lA1[p] = lds_addr(&sA[p][r1 * ASTR]) + ch16;
        lB0[p] = lds_addr(&sB[p][r0 * ASTR]) + ch16;
        lB1[p] = lds_addr(&sB[p][r1 * ASTR]) + ch16;
    }

    auto issue_stage = [&](int k0, int p) {
        async_copy_b128(lA0[p], A + (size_t)(rowBase + r0) * lda + k0 + ch8);
        async_copy_b128(lA1[p], A + (size_t)(rowBase + r1) * lda + k0 + ch8);
        async_copy_b128(lB0[p], Bt + (size_t)bn0c * K + k0 + ch8);
        async_copy_b128(lB1[p], Bt + (size_t)bn1c * K + k0 + ch8);
    };

    v8f acc[2][4] = {};

    // prologue: stage k=0 into buffer 0
    issue_stage(0, 0);
    wait_asynccnt0();
    __syncthreads();

    int p = 0;
    for (int k0 = 0; k0 < K; k0 += BKT) {
        const int kn = k0 + BKT;
        if (kn < K) {
            issue_stage(kn, p ^ 1);             // DMA overlaps the math below
            __builtin_prefetch(A + (size_t)(rowBase + r0) * lda + kn + BKT, 0, 1);
        }

        // ---- fragments (ISA 7.12.2 16-bit layouts) ----
        Frag16 a0, a1, b[4];
        const unsigned* ra0 = (const unsigned*)&sA[p][(wr + fm) * ASTR];
        const unsigned* ra1 = (const unsigned*)&sA[p][(wr + 16 + fm) * ASTR];
#pragma unroll
        for (int q = 0; q < 8; ++q) {
            // A: lane half selects K+8; VGPR q holds K pairs {0,2,4,6,16,18,20,22}
            int kk = (q < 4 ? 2 * q : 2 * (q - 4) + 16) + half * 8;
            a0.u[q] = ra0[kk >> 1];
            a1.u[q] = ra1[kk >> 1];
        }
#pragma unroll
        for (int j = 0; j < 4; ++j) {
            // B: lanes hold N; same K order as A (half selects K+8; pairs {0,2,4,6,16,18,20,22})
            const unsigned* rb = (const unsigned*)&sB[p][(wc + j * 16 + fm) * ASTR];
#pragma unroll
            for (int q = 0; q < 8; ++q) b[j].u[q] = rb[(q < 4 ? q : q + 4) + half * 4];
        }
#pragma unroll
        for (int j = 0; j < 4; ++j) {
            acc[0][j] = wmma_bf16(a0, b[j], acc[0][j]);
            acc[1][j] = wmma_bf16(a1, b[j], acc[1][j]);
        }

        if (kn >= K) break;
        __syncthreads();        // all waves done reading buffer p
        wait_asynccnt0();       // own stage-(k+1) copies have landed
        __syncthreads();        // everyone's copies have landed
        p ^= 1;
    }

    // ---- epilogue (C/D layout: lane = N, VGPR r = M + 8*half) ----
    const int mh = half * 8;
#pragma unroll
    for (int ti = 0; ti < 2; ++ti) {
#pragma unroll
        for (int tj = 0; tj < 4; ++tj) {
            int gcol = colBase + wc + tj * 16 + fm;
            if (gcol >= N) continue;
            float bv = (flags & 1) ? bias[gcol] : 0.0f;
#pragma unroll
            for (int r = 0; r < 8; ++r) {
                int grow = rowBase + wr + ti * 16 + mh + r;
                float v = acc[ti][tj][r] + bv;
                if (flags & 2) v = softplus_f(v);
                if (Cf)  Cf [(size_t)grow * N + gcol] = v;
                if (Cbf) Cbf[(size_t)grow * N + gcol] = (__bf16)v;
            }
        }
    }
}

// ---------------------------------------------------------------------------
// Selective scan: one thread per (b,d); h[16] in registers; fused D-skip and
// SiLU(z) gating; emits bf16 (only consumed by out_proj GEMM).
// ---------------------------------------------------------------------------
__global__ void __launch_bounds__(128)
scan_kernel(const float* __restrict__ delta, const float* __restrict__ xs,
            const float* __restrict__ xz,    const float* __restrict__ proj,
            const float* __restrict__ A_log, const float* __restrict__ Dp,
            __bf16* __restrict__ y_bf) {
    int td = blockIdx.x * 128 + threadIdx.x;           // 0..8191
    int b  = td >> 11;
    int d  = td & (D_INNER - 1);

    float Ad[D_STATE];
#pragma unroll
    for (int n = 0; n < D_STATE; ++n) Ad[n] = -__expf(A_log[d * D_STATE + n]);
    const float Dv = Dp[d];

    float h[D_STATE];
#pragma unroll
    for (int n = 0; n < D_STATE; ++n) h[n] = 0.0f;

    for (int l = 0; l < SEQ; ++l) {
        int t = (b << 10) + l;
        float dv = delta[(size_t)t * D_INNER + d];
        float xv = xs[(size_t)t * D_INNER + d];
        const float4* Bp = (const float4*)(proj + (size_t)t * 96 + DT_RANK);
        const float4* Cp = (const float4*)(proj + (size_t)t * 96 + DT_RANK + D_STATE);
        float accv = 0.0f;
#pragma unroll
        for (int q = 0; q < 4; ++q) {
            float4 Bv = Bp[q];
            float4 Cv = Cp[q];
            float dbx = dv * xv;
            int n = 4 * q;
            h[n+0] = __expf(dv*Ad[n+0])*h[n+0] + dbx*Bv.x; accv += h[n+0]*Cv.x;
            h[n+1] = __expf(dv*Ad[n+1])*h[n+1] + dbx*Bv.y; accv += h[n+1]*Cv.y;
            h[n+2] = __expf(dv*Ad[n+2])*h[n+2] + dbx*Bv.z; accv += h[n+2]*Cv.z;
            h[n+3] = __expf(dv*Ad[n+3])*h[n+3] + dbx*Bv.w; accv += h[n+3]*Cv.w;
        }
        float zv = xz[(size_t)t * (2 * D_INNER) + D_INNER + d];
        y_bf[(size_t)t * D_INNER + d] = (__bf16)((accv + xv * Dv) * silu_f(zv));
    }
}

// ---------------------------------------------------------------------------
// Host-side orchestration
// ---------------------------------------------------------------------------
extern "C" void kernel_launch(void* const* d_in, const int* in_sizes, int n_in,
                              void* d_out, int out_size, void* d_ws, size_t ws_size,
                              hipStream_t stream) {
    (void)in_sizes; (void)n_in; (void)out_size; (void)ws_size;

    const int*   src       = (const int*)  d_in[0];
    const float* emb       = (const float*)d_in[1];
    const float* in_proj_w = (const float*)d_in[2];
    const float* conv_w    = (const float*)d_in[3];
    const float* conv_b    = (const float*)d_in[4];
    const float* x_proj_w  = (const float*)d_in[5];
    const float* dt_proj_w = (const float*)d_in[6];
    const float* dt_proj_b = (const float*)d_in[7];
    const float* A_log     = (const float*)d_in[8];
    const float* Dp        = (const float*)d_in[9];
    const float* out_proj_w= (const float*)d_in[10];
    const float* dec_w     = (const float*)d_in[11];
    const float* dec_b     = (const float*)d_in[12];
    float* out = (float*)d_out;

    // ---- workspace carve-up (256B aligned), ~202 MB total ----
    char*  base = (char*)d_ws;
    size_t off  = 0;
    auto alloc = [&](size_t bytes) -> void* {
        void* p = base + off;
        off += (bytes + 255) & ~(size_t)255;
        return p;
    };
    float*  xz      = (float*) alloc((size_t)NTOK * 2 * D_INNER * 4);
    float*  xs      = (float*) alloc((size_t)NTOK * D_INNER * 4);
    float*  proj    = (float*) alloc((size_t)NTOK * 96 * 4);
    float*  delta   = (float*) alloc((size_t)NTOK * D_INNER * 4);
    __bf16* x_bf    = (__bf16*)alloc((size_t)NTOK * D_MODEL * 2);
    __bf16* xs_bf   = (__bf16*)alloc((size_t)NTOK * D_INNER * 2);
    __bf16* proj_bf = (__bf16*)alloc((size_t)NTOK * 96 * 2);
    __bf16* y_bf    = (__bf16*)alloc((size_t)NTOK * D_INNER * 2);
    __bf16* h_bf    = (__bf16*)alloc((size_t)NTOK * D_MODEL * 2);
    __bf16* w_in_t  = (__bf16*)alloc((size_t)D_MODEL * 2 * D_INNER * 2);
    __bf16* w_xp_t  = (__bf16*)alloc((size_t)D_INNER * 96 * 2);
    __bf16* w_dt_t  = (__bf16*)alloc((size_t)DT_RANK * D_INNER * 2);
    __bf16* w_op_t  = (__bf16*)alloc((size_t)D_INNER * D_MODEL * 2);
    __bf16* w_dec_t = (__bf16*)alloc((size_t)D_MODEL * OUT_DIM * 2);

    auto tgrid = [](size_t n) { return (unsigned)((n + 255) / 256); };
    auto cgrid = [](int n)    { return (unsigned)((n + BNT - 1) / BNT); };

    // ---- 0) weight transpose+convert to [N][K] bf16 ----
    transpose_convert_kernel<<<tgrid((size_t)D_MODEL*2*D_INNER), 256, 0, stream>>>(
        in_proj_w, w_in_t, D_MODEL, 2*D_INNER);
    transpose_convert_kernel<<<tgrid((size_t)D_INNER*96), 256, 0, stream>>>(
        x_proj_w, w_xp_t, D_INNER, 96);
    transpose_convert_kernel<<<tgrid((size_t)DT_RANK*D_INNER), 256, 0, stream>>>(
        dt_proj_w, w_dt_t, DT_RANK, D_INNER);
    transpose_convert_kernel<<<tgrid((size_t)D_INNER*D_MODEL), 256, 0, stream>>>(
        out_proj_w, w_op_t, D_INNER, D_MODEL);
    transpose_convert_kernel<<<tgrid((size_t)D_MODEL*OUT_DIM), 256, 0, stream>>>(
        dec_w, w_dec_t, D_MODEL, OUT_DIM);

    // ---- 1) embedding ----
    embed_kernel<<<tgrid((size_t)NTOK*D_MODEL), 256, 0, stream>>>(src, emb, x_bf);

    // ---- 2) xz = x @ in_proj_w ----
    gemm_wmma_async<<<dim3(cgrid(2*D_INNER), NTOK/BMT), 256, 0, stream>>>(
        x_bf, D_MODEL, w_in_t, xz, nullptr, nullptr, NTOK, 2*D_INNER, D_MODEL, 0);

    // ---- 3) depthwise conv + SiLU ----
    conv_silu_kernel<<<tgrid((size_t)NTOK*D_INNER), 256, 0, stream>>>(
        xz, conv_w, conv_b, xs, xs_bf);

    // ---- 4) proj = xs @ x_proj_w  (fp32 for scan B/C + bf16 for dt GEMM) ----
    gemm_wmma_async<<<dim3(cgrid(96), NTOK/BMT), 256, 0, stream>>>(
        xs_bf, D_INNER, w_xp_t, proj, proj_bf, nullptr, NTOK, 96, D_INNER, 0);

    // ---- 5) delta = softplus(dt @ dt_proj_w + b); dt = proj[:,0:64], lda=96 ----
    gemm_wmma_async<<<dim3(cgrid(D_INNER), NTOK/BMT), 256, 0, stream>>>(
        proj_bf, 96, w_dt_t, delta, nullptr, dt_proj_b, NTOK, D_INNER, DT_RANK, 3);

    // ---- 6) selective scan (+D skip, +SiLU(z) gate) ----
    scan_kernel<<<(BATCH * D_INNER)/128, 128, 0, stream>>>(
        delta, xs, xz, proj, A_log, Dp, y_bf);

    // ---- 7) h = y @ out_proj_w  (bf16 only) ----
    gemm_wmma_async<<<dim3(cgrid(D_MODEL), NTOK/BMT), 256, 0, stream>>>(
        y_bf, D_INNER, w_op_t, nullptr, h_bf, nullptr, NTOK, D_MODEL, D_INNER, 0);

    // ---- 8) out = h @ dec_w + dec_b ----
    gemm_wmma_async<<<dim3(cgrid(OUT_DIM), NTOK/BMT), 256, 0, stream>>>(
        h_bf, D_MODEL, w_dec_t, out, nullptr, dec_b, NTOK, OUT_DIM, D_MODEL, 1);
}
